// HeteroSAGE_44968307589308
// MI455X (gfx1250) — matrix operand order repaired and checked by reading, not compiled, not run on hardware
//
#include <hip/hip_runtime.h>
#include <hip/hip_bf16.h>

typedef __attribute__((ext_vector_type(16))) __bf16 v16bf;
typedef __attribute__((ext_vector_type(8)))  __bf16 v8bf;
typedef __attribute__((ext_vector_type(8)))  float  v8f;

#define HD 128

// ---------------- utility kernels ----------------

__global__ void k_zero_f4(float* __restrict__ p, long n4) {
    long t = (long)blockIdx.x * blockDim.x + threadIdx.x;
    if (t < n4) ((float4*)p)[t] = make_float4(0.f, 0.f, 0.f, 0.f);
}

__global__ void k_f32_to_bf16(const float* __restrict__ src, __bf16* __restrict__ dst, int n) {
    int t = blockIdx.x * blockDim.x + threadIdx.x;
    if (t < n) dst[t] = (__bf16)src[t];
}

__global__ void k_relu_copy(const float* __restrict__ Y, float* __restrict__ X, long n4) {
    long t = (long)blockIdx.x * blockDim.x + threadIdx.x;
    if (t < n4) {
        float4 v = ((const float4*)Y)[t];
        v.x = fmaxf(v.x, 0.f); v.y = fmaxf(v.y, 0.f);
        v.z = fmaxf(v.z, 0.f); v.w = fmaxf(v.w, 0.f);
        ((float4*)X)[t] = v;
    }
}

// ---------------- embedding ----------------
// out[n][:] = nte[x0] + cte[ct] + pte[pt]; each thread does 4 columns of one node
__global__ void k_embed(const int* __restrict__ x, int n, int isComp,
                        const float* __restrict__ nte, const float* __restrict__ cte,
                        const float* __restrict__ pte, float* __restrict__ out) {
    int t = blockIdx.x * blockDim.x + threadIdx.x;
    if (t >= n * 32) return;
    int node = t >> 5;
    int h = (t & 31) << 2;
    int nt = x[node * 3 + 0];
    int c1 = x[node * 3 + 1];
    int ct = isComp ? 0 : (c1 > 0 ? c1 : 0);
    int p1 = x[node * 3 + 2];
    int pt = p1 > 0 ? p1 : 0;
    float4 a = *(const float4*)(nte + (size_t)nt * HD + h);
    float4 b = *(const float4*)(cte + (size_t)ct * HD + h);
    float4 c = *(const float4*)(pte + (size_t)pt * HD + h);
    float4 o;
    o.x = a.x + b.x + c.x; o.y = a.y + b.y + c.y;
    o.z = a.z + b.z + c.z; o.w = a.w + b.w + c.w;
    *(float4*)(out + (size_t)node * HD + h) = o;
}

// ---------------- edge scatter (segment-sum + count) ----------------
// one wave per edge: each lane atomically adds 4 consecutive columns
__global__ void k_scatter(const int* __restrict__ ei, int E,
                          const float* __restrict__ Xs,
                          float* __restrict__ agg, float* __restrict__ cnt) {
    int e = blockIdx.x * (blockDim.x >> 5) + (threadIdx.x >> 5);
    if (e >= E) return;
    int lane = threadIdx.x & 31;
    int src = ei[e];
    int dst = ei[E + e];
    float4 v = ((const float4*)(Xs + (size_t)src * HD))[lane];
    float* a = agg + (size_t)dst * HD + lane * 4;
    atomicAdd(a + 0, v.x);
    atomicAdd(a + 1, v.y);
    atomicAdd(a + 2, v.z);
    atomicAdd(a + 3, v.w);
    if (lane == 0) atomicAdd(cnt + dst, 1.0f);
}

// ---------------- per-relation GEMM via WMMA ----------------
// Y_d[tile] += mean @ Wl^T + bl + X_d @ Wr^T
// block = 256 threads = 8 waves; wave w owns N-tile w (H=128 -> 8 tiles of 16)
// grid.x = size_d/16 (exact: all node counts are multiples of 16)
__global__ void k_gemm_rel(const float* __restrict__ agg, const float* __restrict__ cnt,
                           const float* __restrict__ Xd,
                           const __bf16* __restrict__ Wl, const __bf16* __restrict__ Wr,
                           const float* __restrict__ bvec, float* __restrict__ Yd) {
    int lane = threadIdx.x & 31;
    int wv   = threadIdx.x >> 5;     // N tile index 0..7
    int half = lane >> 4;            // 0: lanes 0-15, 1: lanes 16-31
    int l16  = lane & 15;
    int m = blockIdx.x * 16 + l16;   // A row handled by this lane
    int n = wv * 16 + l16;           // B column (= W row) handled by this lane
    float inv = 1.0f / fmaxf(cnt[m], 1.0f);
    const float* arow = agg + (size_t)m * HD;
    const float* xrow = Xd  + (size_t)m * HD;
    const __bf16* wlr = Wl + (size_t)n * HD;
    const __bf16* wrr = Wr + (size_t)n * HD;

    v8f acc = {0.f, 0.f, 0.f, 0.f, 0.f, 0.f, 0.f, 0.f};

#pragma unroll
    for (int k0 = 0; k0 < HD; k0 += 32) {
        // A fragment layout (16-bit A, 16x32): lane half*8 gives K base; elems 0..7 -> K, 8..15 -> K+16
        int ka = k0 + half * 8;
        float4 a0 = *(const float4*)(arow + ka);
        float4 a1 = *(const float4*)(arow + ka + 4);
        float4 a2 = *(const float4*)(arow + ka + 16);
        float4 a3 = *(const float4*)(arow + ka + 20);
        float4 x0 = *(const float4*)(xrow + ka);
        float4 x1 = *(const float4*)(xrow + ka + 4);
        float4 x2 = *(const float4*)(xrow + ka + 16);
        float4 x3 = *(const float4*)(xrow + ka + 20);
        v16bf am, ax;
        am[0]  = (__bf16)(a0.x * inv); am[1]  = (__bf16)(a0.y * inv);
        am[2]  = (__bf16)(a0.z * inv); am[3]  = (__bf16)(a0.w * inv);
        am[4]  = (__bf16)(a1.x * inv); am[5]  = (__bf16)(a1.y * inv);
        am[6]  = (__bf16)(a1.z * inv); am[7]  = (__bf16)(a1.w * inv);
        am[8]  = (__bf16)(a2.x * inv); am[9]  = (__bf16)(a2.y * inv);
        am[10] = (__bf16)(a2.z * inv); am[11] = (__bf16)(a2.w * inv);
        am[12] = (__bf16)(a3.x * inv); am[13] = (__bf16)(a3.y * inv);
        am[14] = (__bf16)(a3.z * inv); am[15] = (__bf16)(a3.w * inv);
        ax[0]  = (__bf16)x0.x; ax[1]  = (__bf16)x0.y;
        ax[2]  = (__bf16)x0.z; ax[3]  = (__bf16)x0.w;
        ax[4]  = (__bf16)x1.x; ax[5]  = (__bf16)x1.y;
        ax[6]  = (__bf16)x1.z; ax[7]  = (__bf16)x1.w;
        ax[8]  = (__bf16)x2.x; ax[9]  = (__bf16)x2.y;
        ax[10] = (__bf16)x2.z; ax[11] = (__bf16)x2.w;
        ax[12] = (__bf16)x3.x; ax[13] = (__bf16)x3.y;
        ax[14] = (__bf16)x3.z; ax[15] = (__bf16)x3.w;

        // B fragment layout (16-bit B, 32x16): lane half*8 gives K base; elems 0..7 -> K, 8..15 -> K+16
        int kw = k0 + half * 8;
        v8bf wl0 = *(const v8bf*)(wlr + kw);
        v8bf wl1 = *(const v8bf*)(wlr + kw + 16);
        v8bf wr0 = *(const v8bf*)(wrr + kw);
        v8bf wr1 = *(const v8bf*)(wrr + kw + 16);
        v16bf bwl, bwr;
#pragma unroll
        for (int j = 0; j < 8; ++j) {
            bwl[j] = wl0[j]; bwl[8 + j] = wl1[j];
            bwr[j] = wr0[j]; bwr[8 + j] = wr1[j];
        }

        acc = __builtin_amdgcn_wmma_f32_16x16x32_bf16(false, am, false, bwl, (short)0, acc, false, false);
        acc = __builtin_amdgcn_wmma_f32_16x16x32_bf16(false, ax, false, bwr, (short)0, acc, false, false);
    }

    // C/D layout: VGPR r, lanes 0-15 -> M=r, lanes 16-31 -> M=r+8; N = l16
    float bias = bvec[n];
    int rowb = blockIdx.x * 16 + half * 8;
    int col  = wv * 16 + l16;
#pragma unroll
    for (int r = 0; r < 8; ++r) {
        float* y = Yd + (size_t)(rowb + r) * HD + col;
        *y += acc[r] + bias;
    }
}

// ---------------- classifier: out = Xc @ Wc^T + bc  (N=10, tiny) ----------------
__global__ void k_classify(const float* __restrict__ X, const float* __restrict__ Wc,
                           const float* __restrict__ bc, float* __restrict__ out, int nNodes) {
    int t = blockIdx.x * blockDim.x + threadIdx.x;
    if (t >= nNodes * 10) return;
    int node = t / 10;
    int c = t - node * 10;
    const float4* xr = (const float4*)(X + (size_t)node * HD);
    const float4* wr = (const float4*)(Wc + (size_t)c * HD);
    float s = 0.f;
#pragma unroll
    for (int j = 0; j < 32; ++j) {
        float4 a = xr[j], b = wr[j];
        s += a.x * b.x + a.y * b.y + a.z * b.z + a.w * b.w;
    }
    out[t] = s + bc[c];
}

// ---------------- host orchestration ----------------

extern "C" void kernel_launch(void* const* d_in, const int* in_sizes, int n_in,
                              void* d_out, int out_size, void* d_ws, size_t ws_size,
                              hipStream_t stream) {
    const int H = 128;
    const int NCn = 100000, NPn = 400000, NSn = 10000, NNn = 120000;
    const int nsz[4] = {NCn, NPn, NSn, NNn};
    const size_t nX = (size_t)(NCn + NPn + NSn + NNn) * H;   // 80,640,000 floats

    float* ws = (float*)d_ws;
    float* Xarr[4];
    Xarr[0] = ws;
    Xarr[1] = Xarr[0] + (size_t)NCn * H;
    Xarr[2] = Xarr[1] + (size_t)NPn * H;
    Xarr[3] = Xarr[2] + (size_t)NSn * H;
    float* Yb = ws + nX;
    float* Yarr[4];
    Yarr[0] = Yb;
    Yarr[1] = Yarr[0] + (size_t)NCn * H;
    Yarr[2] = Yarr[1] + (size_t)NPn * H;
    Yarr[3] = Yarr[2] + (size_t)NSn * H;
    float* agg = Yb + nX;                       // sized for largest dst (pin)
    float* cnt = agg + (size_t)NPn * H;
    __bf16* Wb = (__bf16*)(cnt + NPn);          // byte offset is a multiple of 16
    const int WN = 6 * 128 * 128;               // elems per weight tensor
    __bf16* Wlb[2] = { Wb,          Wb + 2 * (size_t)WN };
    __bf16* Wrb[2] = { Wb + WN,     Wb + 3 * (size_t)WN };

    // 1) weights -> bf16 (Wl1=13, Wr1=15, Wl2=16, Wr2=18)
    k_f32_to_bf16<<<(WN + 255) / 256, 256, 0, stream>>>((const float*)d_in[13], Wlb[0], WN);
    k_f32_to_bf16<<<(WN + 255) / 256, 256, 0, stream>>>((const float*)d_in[15], Wrb[0], WN);
    k_f32_to_bf16<<<(WN + 255) / 256, 256, 0, stream>>>((const float*)d_in[16], Wlb[1], WN);
    k_f32_to_bf16<<<(WN + 255) / 256, 256, 0, stream>>>((const float*)d_in[18], Wrb[1], WN);

    // 2) embeddings (node_type_emb=10, comp_type_emb=11, pin_type_emb=12)
    const float* nte = (const float*)d_in[10];
    const float* cte = (const float*)d_in[11];
    const float* pte = (const float*)d_in[12];
    for (int t = 0; t < 4; ++t) {
        int nthr = nsz[t] * 32;
        k_embed<<<(nthr + 255) / 256, 256, 0, stream>>>(
            (const int*)d_in[t], nsz[t], (t == 0) ? 1 : 0, nte, cte, pte, Xarr[t]);
    }

    // relations: edge-input index, src type, dst type
    const int relEin[6] = {4, 5, 6, 7, 8, 9};
    const int relS[6]   = {0, 1, 2, 1, 1, 3};
    const int relD[6]   = {1, 0, 1, 2, 3, 1};
    const float* blp[2] = {(const float*)d_in[14], (const float*)d_in[17]};

    // 3) two hetero layers
    for (int L = 0; L < 2; ++L) {
        long y4 = (long)(nX / 4);
        k_zero_f4<<<(unsigned)((y4 + 255) / 256), 256, 0, stream>>>(Yb, y4);
        for (int i = 0; i < 6; ++i) {
            int sd = nsz[relD[i]];
            int E = in_sizes[relEin[i]] / 2;
            long a4 = (long)sd * H / 4;
            k_zero_f4<<<(unsigned)((a4 + 255) / 256), 256, 0, stream>>>(agg, a4);
            long c4 = (long)sd / 4;
            k_zero_f4<<<(unsigned)((c4 + 255) / 256), 256, 0, stream>>>(cnt, c4);
            k_scatter<<<(E + 7) / 8, 256, 0, stream>>>(
                (const int*)d_in[relEin[i]], E, Xarr[relS[i]], agg, cnt);
            k_gemm_rel<<<sd / 16, 256, 0, stream>>>(
                agg, cnt, Xarr[relD[i]],
                Wlb[L] + (size_t)i * 16384, Wrb[L] + (size_t)i * 16384,
                blp[L] + (size_t)i * H, Yarr[relD[i]]);
        }
        // relu(Y) -> X for next stage
        k_relu_copy<<<(unsigned)((y4 + 255) / 256), 256, 0, stream>>>(Yb, ws, y4);
    }

    // 4) classifier (Wc=19, bc=20)
    int nt = NCn * 10;
    k_classify<<<(nt + 255) / 256, 256, 0, stream>>>(
        Xarr[0], (const float*)d_in[19], (const float*)d_in[20], (float*)d_out, NCn);
}
